// AttentionalJoin_49692771615201
// MI455X (gfx1250) — matrix operand order repaired and checked by reading, not compiled, not run on hardware
//
#include <hip/hip_runtime.h>
#include <hip/hip_bf16.h>
#include <math.h>

// ---------------------------------------------------------------------------
// Problem constants (from reference): B=16, N=2048, M=2049, C=512, H=8, hd=64
// ---------------------------------------------------------------------------
#define BATCH   16
#define SEQ     2048
#define MTOK    2049          // N + CLS
#define MPAD    2176          // 17 * 128 (padded token rows per batch)
#define CDIM    512
#define NKV     1024          // K(512) + V(512) output features
#define HEADS   8
#define HD      64

// workspace layout (bytes)
#define TOK_OFF   0ull                                  // 16*2176*512*2 = 35,651,584
#define WKV_OFF   35651584ull                           // 1024*512*2    =  1,048,576
#define KV_OFF    36700160ull                           // 16*2049*1024*2= 67,141,632
#define QCLS_OFF  103841792ull                          // 512*4
#define ATT_OFF   103843840ull                          // 16*512*4

typedef __attribute__((ext_vector_type(16))) __bf16 v16bf;
typedef __attribute__((ext_vector_type(8)))  float  v8f;

union Frag { uint4 u[2]; v16bf v; };

__device__ __forceinline__ float bf2f(unsigned short s) {
    union { unsigned u; float f; } x; x.u = ((unsigned)s) << 16; return x.f;
}

// Async global->LDS 16-byte copy (CDNA5 ASYNCcnt path).
// vdst = LDS byte offset (AS3 ptrtoint), vaddr = 64-bit global address.
__device__ __forceinline__ void async_copy_b128(const void* gaddr, void* lptr) {
    unsigned loff = (unsigned)(size_t)(__attribute__((address_space(3))) void*)lptr;
    asm volatile("global_load_async_to_lds_b128 %0, %1, off"
                 :: "v"(loff), "v"(gaddr)
                 : "memory");
}

// ---------------------------------------------------------------------------
// Kernel 0: fp32 -> bf16 conversion of tokens (with CLS prepend + zero pad)
//           and of the K/V half of qkv_w (rows 512..1535).
// ---------------------------------------------------------------------------
__global__ __launch_bounds__(256) void aj_convert(
    const float* __restrict__ x, const float* __restrict__ cls,
    const float* __restrict__ qkv_w,
    __hip_bfloat16* __restrict__ tokens, __hip_bfloat16* __restrict__ wkv) {
    const long long T0 = (long long)BATCH * MPAD * CDIM;    // 17,825,792
    const long long T1 = (long long)NKV * CDIM;             //    524,288
    const long long total = T0 + T1;
    long long stride = (long long)gridDim.x * blockDim.x;
    for (long long idx = (long long)blockIdx.x * blockDim.x + threadIdx.x;
         idx < total; idx += stride) {
        if (idx < T0) {
            int c = (int)(idx & (CDIM - 1));
            int r = (int)((idx >> 9) % MPAD);
            int b = (int)(idx / ((long long)MPAD * CDIM));
            float v;
            if (r == 0)            v = cls[c];
            else if (r <= SEQ)     v = x[((long long)b * SEQ + (r - 1)) * CDIM + c];
            else                   v = 0.0f;
            tokens[idx] = __float2bfloat16(v);
        } else {
            long long j = idx - T0;                          // 0 .. 524287
            wkv[j] = __float2bfloat16(qkv_w[(long long)CDIM * CDIM + j]);
        }
    }
}

// ---------------------------------------------------------------------------
// Kernel 1: q_cls[c] = sum_k cls[k] * qkv_w[c, k]   (c in [0, 512))
// ---------------------------------------------------------------------------
__global__ __launch_bounds__(256) void aj_qcls(
    const float* __restrict__ cls, const float* __restrict__ qkv_w,
    float* __restrict__ qcls) {
    __shared__ float sc[CDIM];
    int t = threadIdx.x;
    sc[t]       = cls[t];
    sc[t + 256] = cls[t + 256];
    __syncthreads();
    int c = blockIdx.x * 256 + t;
    const float* wr = qkv_w + (long long)c * CDIM;
    float s = 0.f;
    #pragma unroll 8
    for (int k = 0; k < CDIM; ++k) s += sc[k] * wr[k];
    qcls[c] = s;
}

// ---------------------------------------------------------------------------
// Kernel 2: KV projection GEMM (bf16 WMMA, f32 accum)
//   kv[b, m, n] = sum_k tokens[b, m, k] * wkv[n, k]      m<2049, n<1024
//   Block tile: 128(M) x 128(N) x 32(K); 8 waves, each 64x32 (4x2 WMMA tiles)
//   Staging: double-buffered LDS filled by global_load_async_to_lds_b128,
//   handoff via s_wait_asynccnt + workgroup barrier.
// ---------------------------------------------------------------------------
#define LDSTRIDE 80          // bytes per 32-element bf16 row (64B data + 16B pad)
#define ABYTES   (128 * LDSTRIDE)   // 10240 per operand tile
#define BUFBYTES (2 * ABYTES)       // 20480 per stage buffer

__global__ __launch_bounds__(256) void aj_gemm_kv(
    const __hip_bfloat16* __restrict__ tokens,
    const __hip_bfloat16* __restrict__ wkv,
    __hip_bfloat16* __restrict__ kv) {
    __shared__ __align__(16) unsigned char lds[2 * BUFBYTES]; // 40960 B

    const int b  = blockIdx.z;
    const int bm = blockIdx.y;      // 0..16
    const int bn = blockIdx.x;      // 0..7
    const int tid  = threadIdx.x;
    const int wave = tid >> 5;
    const int lane = tid & 31;
    const int wm = wave & 1;        // 0..1  (64-row strip)
    const int wn = wave >> 1;       // 0..3  (32-col strip)
    const int lr = lane & 15;
    const int lh = lane >> 4;

    // staging: each thread owns 2 x 16B chunks per operand tile
    const uint4* tok4 = (const uint4*)tokens;   // 64 uint4 per 512-elem row
    const uint4* wkv4 = (const uint4*)wkv;

    const int id0 = tid;            // chunk ids 0..511 (row = id>>2, kc = id&3)
    const int id1 = tid + 256;
    const int ar0 = id0 >> 2, ac0 = id0 & 3;
    const int ar1 = id1 >> 2, ac1 = id1 & 3;

    const long long aBase = ((long long)b * MPAD + (long long)bm * 128) * 64;
    const long long bBase = ((long long)bn * 128) * 64;

    const int KT = CDIM / 32;   // 16

    // issue the 4 async copies for k-tile kt into stage buffer `buf`
    auto issue_tile = [&](int kt, unsigned char* buf) {
        const long long ko = (long long)kt * 4;
        async_copy_b128(&tok4[aBase + (long long)ar0 * 64 + ko + ac0],
                        buf + ar0 * LDSTRIDE + ac0 * 16);
        async_copy_b128(&tok4[aBase + (long long)ar1 * 64 + ko + ac1],
                        buf + ar1 * LDSTRIDE + ac1 * 16);
        async_copy_b128(&wkv4[bBase + (long long)ar0 * 64 + ko + ac0],
                        buf + ABYTES + ar0 * LDSTRIDE + ac0 * 16);
        async_copy_b128(&wkv4[bBase + (long long)ar1 * 64 + ko + ac1],
                        buf + ABYTES + ar1 * LDSTRIDE + ac1 * 16);
    };

    issue_tile(0, lds);
    issue_tile(1, lds + BUFBYTES);

    v8f acc[4][2];
    #pragma unroll
    for (int mi = 0; mi < 4; ++mi)
        #pragma unroll
        for (int ni = 0; ni < 2; ++ni)
            acc[mi][ni] = v8f{0.f, 0.f, 0.f, 0.f, 0.f, 0.f, 0.f, 0.f};

    for (int kt = 0; kt < KT; ++kt) {
        // wait for tile kt (its 4 copies were issued first; async loads
        // complete in order, so <=4 outstanding means tile kt is in LDS)
        if (kt + 1 < KT) asm volatile("s_wait_asynccnt 0x4" ::: "memory");
        else             asm volatile("s_wait_asynccnt 0x0" ::: "memory");
        __syncthreads();                       // all waves' copies visible

        unsigned char* As = lds + (kt & 1) * BUFBYTES;
        unsigned char* Bs = As + ABYTES;

        // build fragments (ISA 7.12.2 layouts) and issue 8 WMMAs
        Frag af[4], bf[2];
        #pragma unroll
        for (int mi = 0; mi < 4; ++mi) {
            int r = wm * 64 + mi * 16 + lr;
            af[mi].u[0] = *(const uint4*)(As + r * LDSTRIDE + lh * 16);
            af[mi].u[1] = *(const uint4*)(As + r * LDSTRIDE + 32 + lh * 16);
        }
        #pragma unroll
        for (int ni = 0; ni < 2; ++ni) {
            int r = wn * 32 + ni * 16 + lr;
            bf[ni].u[0] = *(const uint4*)(Bs + r * LDSTRIDE + lh * 16);
            bf[ni].u[1] = *(const uint4*)(Bs + r * LDSTRIDE + 32 + lh * 16);
        }
        #pragma unroll
        for (int mi = 0; mi < 4; ++mi)
            #pragma unroll
            for (int ni = 0; ni < 2; ++ni)
                acc[mi][ni] = __builtin_amdgcn_wmma_f32_16x16x32_bf16(
                    false, af[mi].v, false, bf[ni].v,
                    (short)0, acc[mi][ni], false, false);

        __syncthreads();                       // done reading this buffer
        if (kt + 2 < KT) issue_tile(kt + 2, lds + (kt & 1) * BUFBYTES);
    }

    // epilogue: D lane map -> row M = lh*8+i, col N = lr   (store bf16)
    #pragma unroll
    for (int mi = 0; mi < 4; ++mi) {
        #pragma unroll
        for (int ni = 0; ni < 2; ++ni) {
            #pragma unroll
            for (int i = 0; i < 8; ++i) {
                int m = bm * 128 + wm * 64 + mi * 16 + lh * 8 + i;
                if (m < MTOK) {
                    int n = bn * 128 + wn * 32 + ni * 16 + lr;
                    kv[((long long)b * MTOK + m) * NKV + n] =
                        __float2bfloat16(acc[mi][ni][i]);
                }
            }
        }
    }
}

// ---------------------------------------------------------------------------
// Kernel 3: single-query attention per (b, h):
//   logits_j = 0.125 * q_cls[h] . k[b, j, h];  softmax;  out = sum w_j v[b,j,h]
// ---------------------------------------------------------------------------
__global__ __launch_bounds__(256) void aj_attn(
    const __hip_bfloat16* __restrict__ kv, const float* __restrict__ qcls,
    float* __restrict__ attout) {
    __shared__ float w[MTOK];
    __shared__ float red[256];
    __shared__ float part[256];
    __shared__ float qv[HD];

    const int bh = blockIdx.x;
    const int b = bh >> 3, h = bh & 7;
    const int t = threadIdx.x;

    if (t < HD) qv[t] = qcls[h * HD + t];
    __syncthreads();

    // logits
    for (int j = t; j < MTOK; j += 256) {
        const uint4* kr = (const uint4*)(kv + ((long long)b * MTOK + j) * NKV + h * HD);
        float d = 0.f;
        #pragma unroll
        for (int c = 0; c < 8; ++c) {
            uint4 u = kr[c];
            unsigned uu[4] = {u.x, u.y, u.z, u.w};
            #pragma unroll
            for (int q = 0; q < 4; ++q) {
                d += qv[c * 8 + q * 2]     * bf2f((unsigned short)(uu[q] & 0xffffu));
                d += qv[c * 8 + q * 2 + 1] * bf2f((unsigned short)(uu[q] >> 16));
            }
        }
        w[j] = d * 0.125f;   // 1/sqrt(64)
    }
    // block max
    float lm = -3.4e38f;
    for (int j = t; j < MTOK; j += 256) lm = fmaxf(lm, w[j]);
    red[t] = lm; __syncthreads();
    for (int s = 128; s > 0; s >>= 1) {
        if (t < s) red[t] = fmaxf(red[t], red[t + s]);
        __syncthreads();
    }
    float mx = red[0];
    __syncthreads();
    // exp + sum
    float ls = 0.f;
    for (int j = t; j < MTOK; j += 256) {
        float e = __expf(w[j] - mx);
        w[j] = e; ls += e;
    }
    red[t] = ls; __syncthreads();
    for (int s = 128; s > 0; s >>= 1) {
        if (t < s) red[t] += red[t + s];
        __syncthreads();
    }
    float S = red[0];
    __syncthreads();

    // weighted V sum: 4 key-groups x 64 dims
    const int grp = t >> 6;      // 0..3
    const int d   = t & 63;      // 0..63
    float acc = 0.f;
    for (int j = grp; j < MTOK; j += 4) {
        const __hip_bfloat16* vr =
            kv + ((long long)b * MTOK + j) * NKV + CDIM + h * HD;
        acc += w[j] * __bfloat162float(vr[d]);
    }
    part[t] = acc;
    __syncthreads();
    if (grp == 0) {
        float o = (part[d] + part[64 + d] + part[128 + d] + part[192 + d]) / S;
        attout[b * CDIM + h * HD + d] = o;
    }
}

// ---------------------------------------------------------------------------
// Kernel 4: out[b, c] = sum_k attout[b, k] * proj_w[c, k] + proj_b[c]
// ---------------------------------------------------------------------------
__global__ __launch_bounds__(256) void aj_proj(
    const float* __restrict__ attout, const float* __restrict__ proj_w,
    const float* __restrict__ proj_b, float* __restrict__ out) {
    __shared__ float sa[CDIM];
    const int b = blockIdx.x;
    const int t = threadIdx.x;
    sa[t]       = attout[b * CDIM + t];
    sa[t + 256] = attout[b * CDIM + t + 256];
    __syncthreads();
    #pragma unroll
    for (int rep = 0; rep < 2; ++rep) {
        int c = t + rep * 256;
        const float* wr = proj_w + (long long)c * CDIM;
        float s = proj_b[c];
        #pragma unroll 8
        for (int k = 0; k < CDIM; ++k) s += sa[k] * wr[k];
        out[b * CDIM + c] = s;
    }
}

// ---------------------------------------------------------------------------
extern "C" void kernel_launch(void* const* d_in, const int* in_sizes, int n_in,
                              void* d_out, int out_size, void* d_ws, size_t ws_size,
                              hipStream_t stream) {
    const float* x      = (const float*)d_in[0];
    const float* cls    = (const float*)d_in[1];
    const float* qkv_w  = (const float*)d_in[2];
    const float* proj_w = (const float*)d_in[3];
    const float* proj_b = (const float*)d_in[4];
    float* out = (float*)d_out;

    char* ws = (char*)d_ws;
    __hip_bfloat16* tokens = (__hip_bfloat16*)(ws + TOK_OFF);
    __hip_bfloat16* wkv    = (__hip_bfloat16*)(ws + WKV_OFF);
    __hip_bfloat16* kv     = (__hip_bfloat16*)(ws + KV_OFF);
    float* qcls            = (float*)(ws + QCLS_OFF);
    float* attout          = (float*)(ws + ATT_OFF);

    aj_convert<<<8192, 256, 0, stream>>>(x, cls, qkv_w, tokens, wkv);
    aj_qcls<<<2, 256, 0, stream>>>(cls, qkv_w, qcls);
    dim3 g(NKV / 128, MPAD / 128, BATCH);   // (8, 17, 16)
    aj_gemm_kv<<<g, 256, 0, stream>>>(tokens, wkv, kv);
    aj_attn<<<BATCH * HEADS, 256, 0, stream>>>(kv, qcls, attout);
    aj_proj<<<BATCH, 256, 0, stream>>>(attout, proj_w, proj_b, out);
}
